// ClassificationModel_59493886984336
// MI455X (gfx1250) — matrix operand order repaired and checked by reading, not compiled, not run on hardware
//
#include <hip/hip_runtime.h>
#include <hip/hip_bf16.h>

typedef _Float16 f16;
typedef _Float16 h8   __attribute__((ext_vector_type(8)));
typedef _Float16 v16h __attribute__((ext_vector_type(16)));
typedef float    v8f  __attribute__((ext_vector_type(8)));

#define BB   16
#define FF   512
#define TT   32
#define MM   128
#define HH   128
#define CC   61
#define NSEQ (BB * FF)          /* 8192 */
#define ROWS_RNN (NSEQ * TT)    /* 262144 */

// ---------------------------------------------------------------------------
// Fast activations on the recurrent critical path: hardware v_exp_f32 (exp2)
// and v_rcp_f32 instead of branchy libm tanhf/expf.
// ---------------------------------------------------------------------------
__device__ __forceinline__ float fast_sigmoid(float x) {
  return __builtin_amdgcn_rcpf(1.f + __builtin_amdgcn_exp2f(-1.44269504f * x));
}
__device__ __forceinline__ float fast_tanh(float x) {
  // tanh(x) = 1 - 2/(exp(2x)+1)
  return 1.f - 2.f * __builtin_amdgcn_rcpf(
                         1.f + __builtin_amdgcn_exp2f(2.88539008f * x));
}

// ---------------------------------------------------------------------------
// Fragment helpers (CDNA5 wave32 WMMA layouts, 16x16x32 f16)
// ---------------------------------------------------------------------------
__device__ __forceinline__ v16h make_v16(h8 lo, h8 hi) {
  v16h a;
#pragma unroll
  for (int i = 0; i < 8; ++i) { a[i] = lo[i]; a[8 + i] = hi[i]; }
  return a;
}

// B fragment: W is [N][K] row-major f16 (PyTorch weight). B[k][n] = W[n][k].
__device__ __forceinline__ v16h load_bfrag(const f16* __restrict__ W, int K,
                                           int col0, int kt, int lane) {
  const int n = lane & 15;
  const int kBase = (lane < 16) ? 0 : 16;
  const f16* p = W + (size_t)(col0 + n) * K + kt * 32 + kBase;
  h8 lo = *(const h8*)p;
  h8 hi = *(const h8*)(p + 8);
  return make_v16(lo, hi);
}

// A fragment from an LDS f16 tile [16][Kld].
__device__ __forceinline__ v16h load_afrag_lds(const f16* __restrict__ Abuf,
                                               int Kld, int kt, int lane) {
  const int m = lane & 15;
  const int kOff = (lane < 16) ? 0 : 16;
  const f16* p = Abuf + m * Kld + kt * 32 + kOff;
  h8 lo = *(const h8*)p;
  h8 hi = *(const h8*)(p + 8);
  return make_v16(lo, hi);
}

// ---------------------------------------------------------------------------
// f32 -> f16 weight conversion
// ---------------------------------------------------------------------------
__global__ void cvt_w(const float* __restrict__ src, f16* __restrict__ dst,
                      size_t n) {
  size_t i = (size_t)blockIdx.x * blockDim.x + threadIdx.x;
  const size_t stride = (size_t)gridDim.x * blockDim.x;
  for (; i < n; i += stride) dst[i] = (f16)src[i];
}

// ---------------------------------------------------------------------------
// Gate pre-compute GEMM:  G = A @ W^T + bih (+ bhh)
// Logical row = t*S + s; A element = A[s*sStride + t*tStride + k].
// Output tiled f16: G[dir][rowTile][n][m16] (C-fragment layout).
// f16 inputs are staged into LDS with GLOBAL_LOAD_ASYNC_TO_LDS_B128
// (ASYNCcnt-tracked per-lane gather); f32 input (x) converts through VGPRs.
// ---------------------------------------------------------------------------
template <bool F32IN>
__global__ __launch_bounds__(256) void gates_gemm(
    const void* __restrict__ Ain, const f16* __restrict__ Wall,
    const float* __restrict__ bih, const float* __restrict__ bhh,
    f16* __restrict__ Gout, int S, size_t sStride, size_t tStride, size_t rows,
    int K, int N) {
  __shared__ f16 Atile[16 * 256];
  const int tid = threadIdx.x;
  const int lane = tid & 31, wv = tid >> 5;
  const int dir = blockIdx.z;
  const size_t rowTile = blockIdx.x;
  const size_t row0 = rowTile * 16;
  const int s0 = (int)(row0 % S);
  const size_t t = row0 / S;

  const f16* W = Wall + (size_t)dir * N * K;
  f16* G = Gout + (size_t)dir * rows * N;

  if (F32IN) {
    const float* A = (const float*)Ain;
    const int total = 16 * K;
    for (int i = tid; i < total; i += blockDim.x) {
      const int r = i / K, c = i - r * K;
      Atile[r * K + c] = (f16)A[(size_t)(s0 + r) * sStride + t * tStride + c];
    }
  } else {
    // Async global->LDS gather: one 16-byte chunk (8 halves) per lane.
    const f16* A = (const f16*)Ain;
    const int chunksPerRow = K >> 3;          // 32 for K=256
    const int nChunks = 16 * chunksPerRow;    // multiple of 256
    for (int i = tid; i < nChunks; i += blockDim.x) {
      const int r = i / chunksPerRow;
      const int c8 = i - r * chunksPerRow;
      const unsigned ldsOff = (unsigned)((r * K + c8 * 8) * 2);  // Atile @ LDS 0
      const f16* gp = A + (size_t)(s0 + r) * sStride + t * tStride + c8 * 8;
      asm volatile("global_load_async_to_lds_b128 %0, %1, off"
                   :
                   : "v"(ldsOff), "v"((unsigned long long)(uintptr_t)gp)
                   : "memory");
    }
    asm volatile("s_wait_asynccnt 0x0" ::: "memory");
  }
  __syncthreads();

  const int nTile = blockIdx.y * 8 + wv;
  const int col0 = nTile * 16;
  if (col0 >= N) return;
  const int n = lane & 15;
  const int mBase = (lane < 16) ? 0 : 8;

  float bv = bih[dir * N + col0 + n];
  if (bhh) bv += bhh[dir * N + col0 + n];
  v8f acc;
#pragma unroll
  for (int i = 0; i < 8; ++i) acc[i] = bv;

  const int ktc = K >> 5;
  for (int kt = 0; kt < ktc; ++kt) {
    v16h a = load_afrag_lds(Atile, K, kt, lane);
    v16h b = load_bfrag(W, K, col0, kt, lane);
    acc = __builtin_amdgcn_wmma_f32_16x16x32_f16(false, a, false, b, (short)0,
                                                 acc, false, false);
  }

  h8 st;
#pragma unroll
  for (int i = 0; i < 8; ++i) st[i] = (f16)acc[i];
  f16* gp = G + ((size_t)rowTile * N + col0) * 16 + n * 16 + mBase;
  *(h8*)gp = st;
}

// ---------------------------------------------------------------------------
// RNN recurrence: h_t = tanh( G[t] + h_{t-1} @ Whh^T ).
// Whh B-fragments stay in VGPRs for the whole T loop; h ping-pongs in LDS.
// ---------------------------------------------------------------------------
__global__ __launch_bounds__(256) void rnn_recur(
    const f16* __restrict__ G,       // [2][T*S/16 tiles][128][16]
    const f16* __restrict__ WhhAll,  // [2][128][128] f16
    f16* __restrict__ Y,             // [NSEQ][T][256] f16
    int S) {
  __shared__ f16 hbuf[2][16 * 128];
  const int tid = threadIdx.x, lane = tid & 31, wv = tid >> 5;
  const int dir = blockIdx.y;
  const int sTile = blockIdx.x;
  const f16* Whh = WhhAll + (size_t)dir * 128 * 128;
  const f16* Gd = G + (size_t)dir * ((size_t)S * TT) * 128;
  const int col0 = wv * 16;
  const int n = lane & 15;
  const int mBase = (lane < 16) ? 0 : 8;

  v16h bw[4];
#pragma unroll
  for (int kt = 0; kt < 4; ++kt) bw[kt] = load_bfrag(Whh, 128, col0, kt, lane);

  for (int i = tid; i < 16 * 128; i += blockDim.x) hbuf[0][i] = (f16)0.f;
  __syncthreads();

  int cur = 0;
  for (int t = 0; t < TT; ++t) {
    const int tIdx = dir ? (TT - 1 - t) : t;
    const size_t tileIdx = (size_t)tIdx * (S / 16) + sTile;
    const f16* gp = Gd + (tileIdx * 128 + col0) * 16 + n * 16 + mBase;
    h8 gv = *(const h8*)gp;
    v8f acc;
#pragma unroll
    for (int i = 0; i < 8; ++i) acc[i] = (float)gv[i];

#pragma unroll
    for (int kt = 0; kt < 4; ++kt) {
      v16h a = load_afrag_lds(hbuf[cur], 128, kt, lane);
      acc = __builtin_amdgcn_wmma_f32_16x16x32_f16(false, a, false, bw[kt],
                                                   (short)0, acc, false, false);
    }

    const int nxt = cur ^ 1;
#pragma unroll
    for (int i = 0; i < 8; ++i) {
      const f16 hh = (f16)fast_tanh(acc[i]);
      const int m = mBase + i;
      hbuf[nxt][m * 128 + col0 + n] = hh;
      const size_t sG = (size_t)sTile * 16 + m;
      Y[(sG * TT + tIdx) * 256 + dir * 128 + col0 + n] = hh;
    }
    __syncthreads();
    cur = nxt;
  }
}

// ---------------------------------------------------------------------------
// frames = Y2[:, T-1, :]
// ---------------------------------------------------------------------------
__global__ void gather_frames(const f16* __restrict__ Y2,
                              f16* __restrict__ frames) {
  size_t i = (size_t)blockIdx.x * blockDim.x + threadIdx.x;
  if (i >= (size_t)NSEQ * 256) return;
  const size_t s = i >> 8;
  const int c = (int)(i & 255);
  frames[i] = Y2[(s * TT + (TT - 1)) * 256 + c];
}

// ---------------------------------------------------------------------------
// GRU recurrence. Batch = 16 = one WMMA M-tile; 12 WMMAs/step/wave.
// ---------------------------------------------------------------------------
__global__ __launch_bounds__(256) void gru_recur(
    const f16* __restrict__ Gi,        // [2][F tiles][384][16]
    const f16* __restrict__ WhhAll,    // [2][384][128] f16
    const float* __restrict__ bhhAll,  // [2][384]
    f16* __restrict__ Yout) {          // [16][F][256] f16
  __shared__ f16 hbuf[2][16 * 128];
  const int tid = threadIdx.x, lane = tid & 31, wv = tid >> 5;
  const int dir = blockIdx.x;
  const f16* Whh = WhhAll + (size_t)dir * 384 * 128;
  const f16* Gd = Gi + (size_t)dir * (size_t)FF * 384;
  const float* bhh = bhhAll + dir * 384;
  const int col0 = wv * 16;
  const int n = lane & 15;
  const int mBase = (lane < 16) ? 0 : 8;

  v16h bw[3][4];
  float bv[3];
#pragma unroll
  for (int g = 0; g < 3; ++g) {
    bv[g] = bhh[g * 128 + col0 + n];
#pragma unroll
    for (int kt = 0; kt < 4; ++kt)
      bw[g][kt] = load_bfrag(Whh, 128, g * 128 + col0, kt, lane);
  }

  float hreg[8];
#pragma unroll
  for (int i = 0; i < 8; ++i) hreg[i] = 0.f;
  for (int i = tid; i < 16 * 128; i += blockDim.x) hbuf[0][i] = (f16)0.f;
  __syncthreads();

  int cur = 0;
  for (int f = 0; f < FF; ++f) {
    const int fi = dir ? (FF - 1 - f) : f;
    const f16* base = Gd + (size_t)fi * 384 * 16;
    h8 r8 = *(const h8*)(base + (size_t)(0 * 128 + col0) * 16 + n * 16 + mBase);
    h8 z8 = *(const h8*)(base + (size_t)(1 * 128 + col0) * 16 + n * 16 + mBase);
    h8 n8 = *(const h8*)(base + (size_t)(2 * 128 + col0) * 16 + n * 16 + mBase);

    v8f accR, accZ, accN;
    float giN[8];
#pragma unroll
    for (int i = 0; i < 8; ++i) {
      accR[i] = (float)r8[i] + bv[0];
      accZ[i] = (float)z8[i] + bv[1];
      accN[i] = bv[2];
      giN[i] = (float)n8[i];
    }

    v16h a[4];
#pragma unroll
    for (int kt = 0; kt < 4; ++kt)
      a[kt] = load_afrag_lds(hbuf[cur], 128, kt, lane);
#pragma unroll
    for (int kt = 0; kt < 4; ++kt) {
      accR = __builtin_amdgcn_wmma_f32_16x16x32_f16(false, a[kt], false,
                                                    bw[0][kt], (short)0, accR,
                                                    false, false);
      accZ = __builtin_amdgcn_wmma_f32_16x16x32_f16(false, a[kt], false,
                                                    bw[1][kt], (short)0, accZ,
                                                    false, false);
      accN = __builtin_amdgcn_wmma_f32_16x16x32_f16(false, a[kt], false,
                                                    bw[2][kt], (short)0, accN,
                                                    false, false);
    }

    const int nxt = cur ^ 1;
#pragma unroll
    for (int i = 0; i < 8; ++i) {
      const float r = fast_sigmoid(accR[i]);
      const float z = fast_sigmoid(accZ[i]);
      const float nn = fast_tanh(giN[i] + r * accN[i]);
      const float h = (1.f - z) * nn + z * hreg[i];
      hreg[i] = h;
      const f16 hh = (f16)h;
      const int m = mBase + i;  // m == batch index (B == 16)
      hbuf[nxt][m * 128 + col0 + n] = hh;
      Yout[((size_t)m * FF + fi) * 256 + dir * 128 + col0 + n] = hh;
    }
    __syncthreads();
    cur = nxt;
  }
}

// ---------------------------------------------------------------------------
// logits = g2 @ fcW^T + fcb, packed by lengths. (0.13 GFLOP: plain VALU.)
// ---------------------------------------------------------------------------
__global__ void fc_pack(const f16* __restrict__ g2,
                        const float* __restrict__ Wf,
                        const float* __restrict__ bf,
                        const long long* __restrict__ lengths,
                        float* __restrict__ out) {
  const size_t idx = (size_t)blockIdx.x * blockDim.x + threadIdx.x;
  const size_t total = (size_t)BB * FF * CC;
  if (idx >= total) return;
  const int c = (int)(idx % CC);
  const size_t bfi = idx / CC;
  const int f = (int)(bfi % FF);
  const int b = (int)(bfi / FF);
  const long long L = lengths[b];
  if ((long long)f >= L) return;
  long long off = 0;
  for (int i = 0; i < b; ++i) off += lengths[i];
  const f16* xr = g2 + ((size_t)b * FF + f) * 256;
  const float* wr = Wf + (size_t)c * 256;
  float s = bf[c];
  for (int k8 = 0; k8 < 32; ++k8) {
    h8 xv = *(const h8*)(xr + k8 * 8);
#pragma unroll
    for (int j = 0; j < 8; ++j) s += (float)xv[j] * wr[k8 * 8 + j];
  }
  out[(size_t)(off + f) * CC + c] = s;
}

// ---------------------------------------------------------------------------
extern "C" void kernel_launch(void* const* d_in, const int* in_sizes, int n_in,
                              void* d_out, int out_size, void* d_ws,
                              size_t ws_size, hipStream_t stream) {
  (void)in_sizes; (void)n_in; (void)out_size; (void)ws_size;
  const float* x = (const float*)d_in[0];
  char* ws = (char*)d_ws;

  // f16 weight arena (offsets in halves)
  f16* wsW = (f16*)ws;
  const size_t o_r0_Wih = 0;
  const size_t o_r0_Whh = o_r0_Wih + (size_t)2 * 128 * 128;
  const size_t o_r1_Wih = o_r0_Whh + (size_t)2 * 128 * 128;
  const size_t o_r1_Whh = o_r1_Wih + (size_t)2 * 128 * 256;
  const size_t o_g0_Wih = o_r1_Whh + (size_t)2 * 128 * 128;
  const size_t o_g0_Whh = o_g0_Wih + (size_t)2 * 384 * 256;
  const size_t o_g1_Wih = o_g0_Whh + (size_t)2 * 384 * 128;
  const size_t o_g1_Whh = o_g1_Wih + (size_t)2 * 384 * 256;
  const size_t wTotal = o_g1_Whh + (size_t)2 * 384 * 128;

  size_t off = ((wTotal * 2 + 255) / 256) * 256;
  f16* wsG = (f16*)(ws + off);  off += (size_t)2 * ROWS_RNN * 128 * 2;
  f16* wsY1 = (f16*)(ws + off); off += (size_t)NSEQ * TT * 256 * 2;
  f16* wsY2 = (f16*)(ws + off); off += (size_t)NSEQ * TT * 256 * 2;
  f16* wsFr = (f16*)(ws + off); off += (size_t)NSEQ * 256 * 2;
  f16* wsG1 = (f16*)(ws + off); off += (size_t)BB * FF * 256 * 2;
  f16* wsG2 = (f16*)(ws + off); off += (size_t)BB * FF * 256 * 2;

  auto cvt = [&](const void* src, f16* dst, size_t nElem) {
    int blocks = (int)((nElem + 255) / 256);
    if (blocks > 2048) blocks = 2048;
    cvt_w<<<blocks, 256, 0, stream>>>((const float*)src, dst, nElem);
  };
  cvt(d_in[1], wsW + o_r0_Wih, (size_t)2 * 128 * 128);
  cvt(d_in[2], wsW + o_r0_Whh, (size_t)2 * 128 * 128);
  cvt(d_in[5], wsW + o_r1_Wih, (size_t)2 * 128 * 256);
  cvt(d_in[6], wsW + o_r1_Whh, (size_t)2 * 128 * 128);
  cvt(d_in[9], wsW + o_g0_Wih, (size_t)2 * 384 * 256);
  cvt(d_in[10], wsW + o_g0_Whh, (size_t)2 * 384 * 128);
  cvt(d_in[13], wsW + o_g1_Wih, (size_t)2 * 384 * 256);
  cvt(d_in[14], wsW + o_g1_Whh, (size_t)2 * 384 * 128);

  const dim3 blk(256);

  // ---- RNN layer 0 ----
  gates_gemm<true><<<dim3(ROWS_RNN / 16, 1, 2), blk, 0, stream>>>(
      x, wsW + o_r0_Wih, (const float*)d_in[3], (const float*)d_in[4], wsG,
      NSEQ, (size_t)TT * MM, (size_t)MM, (size_t)ROWS_RNN, 128, 128);
  rnn_recur<<<dim3(NSEQ / 16, 2), blk, 0, stream>>>(wsG, wsW + o_r0_Whh, wsY1,
                                                    NSEQ);

  // ---- RNN layer 1 (K = 256) ----
  gates_gemm<false><<<dim3(ROWS_RNN / 16, 1, 2), blk, 0, stream>>>(
      wsY1, wsW + o_r1_Wih, (const float*)d_in[7], (const float*)d_in[8], wsG,
      NSEQ, (size_t)TT * 256, (size_t)256, (size_t)ROWS_RNN, 256, 128);
  rnn_recur<<<dim3(NSEQ / 16, 2), blk, 0, stream>>>(wsG, wsW + o_r1_Whh, wsY2,
                                                    NSEQ);

  // ---- frames = Y2[:, -1, :] ----
  gather_frames<<<(NSEQ * 256 + 255) / 256, 256, 0, stream>>>(wsY2, wsFr);

  // ---- GRU layer 0 ----
  gates_gemm<false><<<dim3(BB * FF / 16, 3, 2), blk, 0, stream>>>(
      wsFr, wsW + o_g0_Wih, (const float*)d_in[11], nullptr, wsG, 16,
      (size_t)FF * 256, (size_t)256, (size_t)BB * FF, 256, 384);
  gru_recur<<<dim3(2), blk, 0, stream>>>(wsG, wsW + o_g0_Whh,
                                         (const float*)d_in[12], wsG1);

  // ---- GRU layer 1 ----
  gates_gemm<false><<<dim3(BB * FF / 16, 3, 2), blk, 0, stream>>>(
      wsG1, wsW + o_g1_Wih, (const float*)d_in[15], nullptr, wsG, 16,
      (size_t)FF * 256, (size_t)256, (size_t)BB * FF, 256, 384);
  gru_recur<<<dim3(2), blk, 0, stream>>>(wsG, wsW + o_g1_Whh,
                                         (const float*)d_in[16], wsG2);

  // ---- FC + length-packing ----
  const size_t total = (size_t)BB * FF * CC;
  fc_pack<<<(int)((total + 255) / 256), 256, 0, stream>>>(
      wsG2, (const float*)d_in[17], (const float*)d_in[18],
      (const long long*)d_in[19], (float*)d_out);
}
